// TriangularAttention_21741124452586
// MI455X (gfx1250) — matrix operand order repaired and checked by reading, not compiled, not run on hardware
//
#include <hip/hip_runtime.h>
#include <hip/hip_bf16.h>

typedef __attribute__((ext_vector_type(16))) __bf16 v16bf;
typedef __attribute__((ext_vector_type(8)))  __bf16 v8bf;
typedef __attribute__((ext_vector_type(8)))  float  v8f;

#define LL 256
#define CC 128
#define HH 4
#define DD 32
#define NROWS (LL * LL)

// A-matrix (16x32 bf16): lane holds row m=lane&15; element e -> K:
//   e=0..15 -> hi*16 + e (contiguous run of 16, same as B)
// B-matrix (32x16 bf16): lane holds col n=lane&15; element e -> K = hi*16 + e
// (contiguous run of 16).  hi = lane>>4.

__device__ __forceinline__ v8f wmma_bf16(v16bf a, v16bf b, v8f c) {
  return __builtin_amdgcn_wmma_f32_16x16x32_bf16(false, a, false, b, (short)0, c,
                                                 false, false);
}

// A-fragment from a bf16 row (LDS or global): two contiguous 8-elem runs.
__device__ __forceinline__ v16bf afrag_bf16(const __bf16* row, int k0, int hi) {
  v8bf lo = *reinterpret_cast<const v8bf*>(row + k0 + hi * 16);
  v8bf hh = *reinterpret_cast<const v8bf*>(row + k0 + hi * 16 + 8);
  return __builtin_shufflevector(lo, hh, 0, 1, 2, 3, 4, 5, 6, 7, 8, 9, 10, 11,
                                 12, 13, 14, 15);
}

// A-fragment converted on the fly from an f32 row.
__device__ __forceinline__ v16bf afrag_f32(const float* row, int k0, int hi) {
  const float4 a0 = *reinterpret_cast<const float4*>(row + k0 + hi * 16);
  const float4 a1 = *reinterpret_cast<const float4*>(row + k0 + hi * 16 + 4);
  const float4 b0 = *reinterpret_cast<const float4*>(row + k0 + hi * 16 + 8);
  const float4 b1 = *reinterpret_cast<const float4*>(row + k0 + hi * 16 + 12);
  v16bf a;
  a[0] = (__bf16)a0.x;  a[1] = (__bf16)a0.y;  a[2] = (__bf16)a0.z;  a[3] = (__bf16)a0.w;
  a[4] = (__bf16)a1.x;  a[5] = (__bf16)a1.y;  a[6] = (__bf16)a1.z;  a[7] = (__bf16)a1.w;
  a[8] = (__bf16)b0.x;  a[9] = (__bf16)b0.y;  a[10] = (__bf16)b0.z; a[11] = (__bf16)b0.w;
  a[12] = (__bf16)b1.x; a[13] = (__bf16)b1.y; a[14] = (__bf16)b1.z; a[15] = (__bf16)b1.w;
  return a;
}

// B-fragment from packed transposed bf16 weights Wt[col][k]: 16 contiguous.
__device__ __forceinline__ v16bf bfrag_packed(const __bf16* Wt, int col, int ks,
                                              int hi) {
  return *reinterpret_cast<const v16bf*>(Wt + (size_t)col * CC + ks * 32 +
                                         hi * 16);
}

// ---------------------------------------------------------------------------
// Kernel 0: pre-pack weights -> bf16, transposed to [col][k].
// pk layout: [Wq|Wk|Wv|Wout|Wgate] (5 x 128*128) then Wb padded to [16][128].
// ---------------------------------------------------------------------------
__global__ __launch_bounds__(256) void k_pack(const float* __restrict__ Wq,
                                              const float* __restrict__ Wk,
                                              const float* __restrict__ Wv,
                                              const float* __restrict__ Wout,
                                              const float* __restrict__ Wgate,
                                              const float* __restrict__ Wb,
                                              __bf16* __restrict__ pk) {
  const int t = blockIdx.x * 256 + threadIdx.x;
  if (t < 5 * CC * CC) {
    const int mat = t / (CC * CC), idx = t % (CC * CC);
    const int col = idx >> 7, k = idx & 127;
    const float* W = (mat == 0)   ? Wq
                     : (mat == 1) ? Wk
                     : (mat == 2) ? Wv
                     : (mat == 3) ? Wout
                                  : Wgate;
    pk[(size_t)mat * CC * CC + idx] = (__bf16)W[(size_t)k * CC + col];
  } else {
    const int idx = t - 5 * CC * CC;  // 0..2047: Wb padded 16x128
    const int col = idx >> 7, k = idx & 127;
    pk[(size_t)5 * CC * CC + idx] =
        (col < HH) ? (__bf16)Wb[(size_t)k * HH + col] : (__bf16)0.0f;
  }
}

// ---------------------------------------------------------------------------
// Kernel 1: LayerNorm + Q/K/V projections (bf16) + pair bias z@Wb
// One wave per 16-row tile of the 65536 flattened pair rows.
// ---------------------------------------------------------------------------
__device__ __forceinline__ void gemm_tile_store(const v16bf a[4],
                                                const __bf16* __restrict__ Wt,
                                                __bf16* __restrict__ out,
                                                int row0, int n, int hi) {
  for (int nt = 0; nt < 8; ++nt) {
    v8f c = {};
#pragma unroll
    for (int ks = 0; ks < 4; ++ks)
      c = wmma_bf16(a[ks], bfrag_packed(Wt, nt * 16 + n, ks, hi), c);
#pragma unroll
    for (int r = 0; r < 8; ++r)
      out[(size_t)(row0 + r + 8 * hi) * CC + nt * 16 + n] = (__bf16)c[r];
  }
}

__global__ __launch_bounds__(32) void k_proj(const float* __restrict__ pair,
                                             const float* __restrict__ ln_g,
                                             const float* __restrict__ ln_b,
                                             const __bf16* __restrict__ pk,
                                             __bf16* __restrict__ qb,
                                             __bf16* __restrict__ kb,
                                             __bf16* __restrict__ vb,
                                             float* __restrict__ biasb) {
  __shared__ alignas(32) __bf16 zt[16][CC];
  const int lane = threadIdx.x;
  const int row0 = blockIdx.x * 16;
  const float4 gv = reinterpret_cast<const float4*>(ln_g)[lane];
  const float4 bv = reinterpret_cast<const float4*>(ln_b)[lane];

  for (int r = 0; r < 16; ++r) {
    const float4 xv =
        reinterpret_cast<const float4*>(pair + (size_t)(row0 + r) * CC)[lane];
    float s = xv.x + xv.y + xv.z + xv.w;
    float sq = xv.x * xv.x + xv.y * xv.y + xv.z * xv.z + xv.w * xv.w;
#pragma unroll
    for (int m = 1; m < 32; m <<= 1) {
      s += __shfl_xor(s, m, 32);
      sq += __shfl_xor(sq, m, 32);
    }
    const float mean = s * (1.0f / 128.0f);
    const float var = sq * (1.0f / 128.0f) - mean * mean;
    const float rstd = rsqrtf(var + 1e-5f);
    zt[r][lane * 4 + 0] = (__bf16)((xv.x - mean) * rstd * gv.x + bv.x);
    zt[r][lane * 4 + 1] = (__bf16)((xv.y - mean) * rstd * gv.y + bv.y);
    zt[r][lane * 4 + 2] = (__bf16)((xv.z - mean) * rstd * gv.z + bv.z);
    zt[r][lane * 4 + 3] = (__bf16)((xv.w - mean) * rstd * gv.w + bv.w);
  }
  __syncthreads();

  const int n = lane & 15, hi = lane >> 4;
  v16bf a[4];
#pragma unroll
  for (int ks = 0; ks < 4; ++ks) a[ks] = afrag_bf16(&zt[n][0], ks * 32, hi);

  gemm_tile_store(a, pk + 0 * CC * CC, qb, row0, n, hi);
  gemm_tile_store(a, pk + 1 * CC * CC, kb, row0, n, hi);
  gemm_tile_store(a, pk + 2 * CC * CC, vb, row0, n, hi);

  // bias = z @ Wb  (packed Wb padded to 16 cols; cols >=4 are zero)
  const __bf16* Wbt = pk + (size_t)5 * CC * CC;
  v8f cb = {};
#pragma unroll
  for (int ks = 0; ks < 4; ++ks)
    cb = wmma_bf16(a[ks], bfrag_packed(Wbt, n, ks, hi), cb);
  if (n < HH) {
#pragma unroll
    for (int r = 0; r < 8; ++r)
      biasb[(size_t)n * NROWS + row0 + r + 8 * hi] = cb[r];
  }
}

// ---------------------------------------------------------------------------
// Kernel 2: fused flash attention per (i, h). 4 waves; each wave owns j-tiles.
// ---------------------------------------------------------------------------
__global__ __launch_bounds__(128) void k_attn(const __bf16* __restrict__ qb,
                                              const __bf16* __restrict__ kb,
                                              const __bf16* __restrict__ vb,
                                              const float* __restrict__ biasb,
                                              __bf16* __restrict__ ob) {
  __shared__ alignas(32) __bf16 Qs[LL * DD];
  __shared__ alignas(32) __bf16 Ks[LL * DD];
  __shared__ alignas(32) __bf16 Vt[DD * LL];  // transposed: Vt[d][k]
  __shared__ alignas(32) __bf16 Pst[4][16 * 32];
  const int i = blockIdx.x, h = blockIdx.y;
  const int tid = threadIdx.x, lane = tid & 31, wave = tid >> 5;
  const int n = lane & 15, hi = lane >> 4;

  for (int idx = tid; idx < LL * 4; idx += 128) {
    const int row = idx >> 2, p = idx & 3;
    const size_t g = ((size_t)(i * LL + row)) * CC + h * DD;
    reinterpret_cast<uint4*>(Qs)[row * 4 + p] =
        reinterpret_cast<const uint4*>(qb + g)[p];
    reinterpret_cast<uint4*>(Ks)[row * 4 + p] =
        reinterpret_cast<const uint4*>(kb + g)[p];
    const v8bf tv = *reinterpret_cast<const v8bf*>(vb + g + p * 8);
#pragma unroll
    for (int u = 0; u < 8; ++u) Vt[(p * 8 + u) * LL + row] = tv[u];
  }
  __syncthreads();

  const float scale = 0.17677669529663687f;  // 1/sqrt(32)
  const float* bh = biasb + (size_t)h * NROWS;

  for (int jt = wave; jt < 16; jt += 4) {
    const v16bf aq = afrag_bf16(Qs + (jt * 16 + n) * DD, 0, hi);

    v8f o0 = {}, o1 = {};
    float mrow[8], lrow[8];
#pragma unroll
    for (int r = 0; r < 8; ++r) {
      mrow[r] = -3.0e38f;
      lrow[r] = 0.0f;
    }

    for (int kt = 0; kt < 8; ++kt) {  // 32 keys per iteration
      const v16bf bk0 =
          *reinterpret_cast<const v16bf*>(Ks + (kt * 32 + n) * DD + hi * 16);
      const v16bf bk1 = *reinterpret_cast<const v16bf*>(
          Ks + (kt * 32 + 16 + n) * DD + hi * 16);
      v8f s0 = {}, s1 = {};
      s0 = wmma_bf16(aq, bk0, s0);
      s1 = wmma_bf16(aq, bk1, s1);

#pragma unroll
      for (int r = 0; r < 8; ++r) {
        const int j = jt * 16 + r + 8 * hi;
        const float v0 = s0[r] * scale + bh[(size_t)j * LL + kt * 32 + n];
        const float v1 = s1[r] * scale + bh[(size_t)j * LL + kt * 32 + 16 + n];
        float mx = fmaxf(v0, v1);
#pragma unroll
        for (int msk = 1; msk < 16; msk <<= 1)
          mx = fmaxf(mx, __shfl_xor(mx, msk, 32));
        const float mnew = fmaxf(mrow[r], mx);
        const float alpha = __expf(mrow[r] - mnew);
        mrow[r] = mnew;
        const float p0 = __expf(v0 - mnew);
        const float p1 = __expf(v1 - mnew);
        lrow[r] = lrow[r] * alpha + (p0 + p1);  // per-lane partial sum
        o0[r] *= alpha;
        o1[r] *= alpha;
        Pst[wave][(r + 8 * hi) * 32 + n] = (__bf16)p0;
        Pst[wave][(r + 8 * hi) * 32 + 16 + n] = (__bf16)p1;
      }
      // wave-private LDS staging; LDS pipe is in-order, just drain stores
      asm volatile("s_wait_dscnt 0" ::: "memory");

      const v16bf ap = afrag_bf16(&Pst[wave][n * 32], 0, hi);
      const v16bf bv0 =
          *reinterpret_cast<const v16bf*>(Vt + n * LL + kt * 32 + hi * 16);
      const v16bf bv1 = *reinterpret_cast<const v16bf*>(Vt + (16 + n) * LL +
                                                        kt * 32 + hi * 16);
      o0 = wmma_bf16(ap, bv0, o0);
      o1 = wmma_bf16(ap, bv1, o1);
    }

#pragma unroll
    for (int r = 0; r < 8; ++r) {
      float l = lrow[r];
#pragma unroll
      for (int msk = 1; msk < 16; msk <<= 1) l += __shfl_xor(l, msk, 32);
      const float invl = 1.0f / l;
      const int j = jt * 16 + r + 8 * hi;
      const size_t g = ((size_t)(i * LL + j)) * CC + h * DD;
      ob[g + n] = (__bf16)(o0[r] * invl);
      ob[g + 16 + n] = (__bf16)(o1[r] * invl);
    }
  }
}

// ---------------------------------------------------------------------------
// Kernel 3: y = O@Wout + bout; gate = sigmoid(pair@Wgate + bgate);
//           out = pair + gate * y.  Pure-register kernel, one wave per tile.
// ---------------------------------------------------------------------------
__global__ __launch_bounds__(32) void k_out(const float* __restrict__ pair,
                                            const __bf16* __restrict__ ob,
                                            const __bf16* __restrict__ pk,
                                            const float* __restrict__ bout,
                                            const float* __restrict__ bgate,
                                            float* __restrict__ outp) {
  const int lane = threadIdx.x;
  const int row0 = blockIdx.x * 16;
  const int n = lane & 15, hi = lane >> 4;
  const __bf16* Woutp = pk + (size_t)3 * CC * CC;
  const __bf16* Wgatep = pk + (size_t)4 * CC * CC;

  const __bf16* orow = ob + (size_t)(row0 + n) * CC;
  const float* prow = pair + (size_t)(row0 + n) * CC;
  v16bf ao[4], ap[4];
#pragma unroll
  for (int ks = 0; ks < 4; ++ks) {
    ao[ks] = afrag_bf16(orow, ks * 32, hi);
    ap[ks] = afrag_f32(prow, ks * 32, hi);
  }

  for (int nt = 0; nt < 8; ++nt) {
    v8f cy = {}, cg = {};
#pragma unroll
    for (int ks = 0; ks < 4; ++ks) {
      cy = wmma_bf16(ao[ks], bfrag_packed(Woutp, nt * 16 + n, ks, hi), cy);
      cg = wmma_bf16(ap[ks], bfrag_packed(Wgatep, nt * 16 + n, ks, hi), cg);
    }
    const float bo = bout[nt * 16 + n];
    const float bgv = bgate[nt * 16 + n];
#pragma unroll
    for (int r = 0; r < 8; ++r) {
      const size_t idx = (size_t)(row0 + r + 8 * hi) * CC + nt * 16 + n;
      const float g = 1.0f / (1.0f + __expf(-(cg[r] + bgv)));
      outp[idx] = pair[idx] + g * (cy[r] + bo);
    }
  }
}

extern "C" void kernel_launch(void* const* d_in, const int* in_sizes, int n_in,
                              void* d_out, int out_size, void* d_ws,
                              size_t ws_size, hipStream_t stream) {
  const float* pair = (const float*)d_in[0];
  const float* ln_g = (const float*)d_in[1];
  const float* ln_b = (const float*)d_in[2];
  const float* Wq = (const float*)d_in[3];
  const float* Wk = (const float*)d_in[4];
  const float* Wv = (const float*)d_in[5];
  const float* Wb = (const float*)d_in[6];
  const float* Wout = (const float*)d_in[7];
  const float* bout = (const float*)d_in[8];
  const float* Wgate = (const float*)d_in[9];
  const float* bgate = (const float*)d_in[10];
  float* out = (float*)d_out;

  char* ws = (char*)d_ws;
  const size_t BF_BYTES = (size_t)NROWS * CC * sizeof(__bf16);  // 16 MiB
  __bf16* qb = (__bf16*)(ws + 0 * BF_BYTES);
  __bf16* kb = (__bf16*)(ws + 1 * BF_BYTES);
  __bf16* vb = (__bf16*)(ws + 2 * BF_BYTES);
  __bf16* ob = (__bf16*)(ws + 3 * BF_BYTES);
  float* biasb = (float*)(ws + 4 * BF_BYTES);        // HH*NROWS f32 = 1 MiB
  __bf16* pk = (__bf16*)(ws + 4 * BF_BYTES + (size_t)HH * NROWS * sizeof(float));

  // pack: 5*128*128 + 16*128 bf16 elements = 83968 -> 328 blocks of 256
  k_pack<<<328, 256, 0, stream>>>(Wq, Wk, Wv, Wout, Wgate, Wb, pk);
  k_proj<<<NROWS / 16, 32, 0, stream>>>(pair, ln_g, ln_b, pk, qb, kb, vb,
                                        biasb);
  dim3 g2(LL, HH);
  k_attn<<<g2, 128, 0, stream>>>(qb, kb, vb, biasb, ob);
  k_out<<<NROWS / 16, 32, 0, stream>>>(pair, ob, pk, bout, bgate, out);
}
